// FMoE_72018011619524
// MI455X (gfx1250) — matrix operand order repaired and checked by reading, not compiled, not run on hardware
//
#include <hip/hip_runtime.h>
#include <math.h>

// Problem constants (match reference)
constexpr int NE = 8;     // experts
constexpr int DM = 1024;  // d_model
constexpr int HH = 2048;  // hidden
constexpr int NT = 2048;  // tokens
constexpr int TOPK = 2;

typedef __attribute__((ext_vector_type(16))) __bf16 v16bf;
typedef __attribute__((ext_vector_type(8)))  float  v8f;

__device__ __forceinline__ unsigned short f2bf(float f) {
  union { float f; unsigned int u; } v; v.f = f;
  unsigned int u = v.u;
  unsigned int r = u + 0x7FFFu + ((u >> 16) & 1u);  // round-to-nearest-even
  return (unsigned short)(r >> 16);
}

__device__ __forceinline__ float gelu_tanh(float x) {
  const float c = 0.7978845608028654f;  // sqrt(2/pi)
  float inner = c * (x + 0.044715f * x * x * x);
  return 0.5f * x * (1.0f + tanhf(inner));
}

// ---------------------------------------------------------------------------
// Zero output + routing counters
// ---------------------------------------------------------------------------
__global__ void zero_kernel(float* out, int n, int* cnt) {
  int i = blockIdx.x * blockDim.x + threadIdx.x;
  if (i < n) out[i] = 0.0f;
  if (i < NE) cnt[i] = 0;
}

// ---------------------------------------------------------------------------
// f32 -> bf16 (bit pattern in ushort)
// ---------------------------------------------------------------------------
__global__ void conv_x_kernel(const float* __restrict__ x,
                              unsigned short* __restrict__ xb, int n) {
  int i = blockIdx.x * blockDim.x + threadIdx.x;
  if (i < n) xb[i] = f2bf(x[i]);
}

// ---------------------------------------------------------------------------
// Per-expert transpose+convert: in [R][C] f32 -> out [C][R] bf16.
// Gives WMMA B-fragments contiguous K per lane.
// ---------------------------------------------------------------------------
__global__ void transconv_kernel(const float* __restrict__ in,
                                 unsigned short* __restrict__ out,
                                 int R, int C) {
  __shared__ float tile[32][33];
  int e = blockIdx.z;
  const float* src = in + (size_t)e * R * C;
  unsigned short* dst = out + (size_t)e * R * C;
  int c0 = blockIdx.x * 32, r0 = blockIdx.y * 32;
  int tx = threadIdx.x, ty = threadIdx.y;  // 32 x 8
  #pragma unroll
  for (int i = 0; i < 32; i += 8)
    tile[ty + i][tx] = src[(size_t)(r0 + ty + i) * C + c0 + tx];
  __syncthreads();
  #pragma unroll
  for (int i = 0; i < 32; i += 8)
    dst[(size_t)(c0 + ty + i) * R + r0 + tx] = f2bf(tile[tx][ty + i]);
}

// ---------------------------------------------------------------------------
// Gate: one wave32 per token. logits = inp @ Wg + bg ; top2 ; softmax(2).
// Appends (token*2+k) to per-expert list; stores gate weight keyed by (t,k).
// ---------------------------------------------------------------------------
__global__ void gate_kernel(const float* __restrict__ inp,
                            const float* __restrict__ Wg,
                            const float* __restrict__ bg,
                            int* cnt, int* list, float* gatew) {
  int wave = threadIdx.x >> 5;
  int lane = threadIdx.x & 31;
  int t = blockIdx.x * 8 + wave;
  if (t >= NT) return;

  float acc[NE];
  #pragma unroll
  for (int e = 0; e < NE; ++e) acc[e] = 0.0f;

  const float* row = inp + (size_t)t * DM;
  for (int d = lane; d < DM; d += 32) {
    float xv = row[d];
    const float* wr = Wg + (size_t)d * NE;
    #pragma unroll
    for (int e = 0; e < NE; ++e) acc[e] += xv * wr[e];
  }
  #pragma unroll
  for (int e = 0; e < NE; ++e) {
    #pragma unroll
    for (int off = 16; off > 0; off >>= 1)
      acc[e] += __shfl_xor(acc[e], off, 32);
  }
  if (lane == 0) {
    float v[NE];
    #pragma unroll
    for (int e = 0; e < NE; ++e) v[e] = acc[e] + bg[e];
    int i0 = 0; float b0 = v[0];
    #pragma unroll
    for (int e = 1; e < NE; ++e) if (v[e] > b0) { b0 = v[e]; i0 = e; }
    int i1 = -1; float b1 = -3.4e38f;
    #pragma unroll
    for (int e = 0; e < NE; ++e)
      if (e != i0 && v[e] > b1) { b1 = v[e]; i1 = e; }
    float e1 = __expf(b1 - b0);
    float w0 = 1.0f / (1.0f + e1);
    float w1 = e1 / (1.0f + e1);
    int p0 = atomicAdd(&cnt[i0], 1);
    list[i0 * NT + p0] = t * 2 + 0;
    gatew[t * 2 + 0] = w0;
    int p1 = atomicAdd(&cnt[i1], 1);
    list[i1 * NT + p1] = t * 2 + 1;
    gatew[t * 2 + 1] = w1;
  }
}

// ---------------------------------------------------------------------------
// Grouped expert GEMM. PHASE 1: h = gelu(x @ W1^T_layout + b1) -> bf16
//                      PHASE 2: out += gate_w * (h @ W2^T_layout + b2) (atomic)
// Block: 256 threads (8 waves), tile M=64 x N=128, K-step 32.
// Wave (mw in 0..1, nw in 0..3) owns a 32x32 tile = 2x2 WMMA 16x16 tiles.
// ---------------------------------------------------------------------------
template <int PHASE>
__global__ __launch_bounds__(256)
void expert_kernel(const unsigned short* __restrict__ Abuf,  // P1: x [NT][DM]; P2: h [NT*2][HH]
                   const unsigned short* __restrict__ Wt,    // P1: [NE][HH][DM]; P2: [NE][DM][HH]
                   const float* __restrict__ bias,           // P1: b1 [NE][HH]; P2: b2 [NE][DM]
                   const int* __restrict__ cnt,
                   const int* __restrict__ list,
                   const float* __restrict__ gatew,
                   unsigned short* __restrict__ hout,
                   float* __restrict__ out) {
  const int KC = (PHASE == 1) ? DM : HH;  // reduction dim
  const int NC = (PHASE == 1) ? HH : DM;  // output columns

  int e = blockIdx.z;
  int count = cnt[e];
  int m_base = blockIdx.y * 64;
  if (m_base >= count) return;
  int n_base = blockIdx.x * 128;

  __shared__ unsigned short sA[64 * 40];   // 64 rows x 32 K, stride 40 (80B, 16B-aligned)
  __shared__ unsigned short sB[128 * 40];  // 128 rows x 32 K

  int tid = threadIdx.x;
  int wave = tid >> 5;
  int lane = tid & 31;
  int mw = wave >> 2;   // 0..1
  int nw = wave & 3;    // 0..3
  int hi = lane >> 4;   // lane group
  int lm = lane & 15;

  const int* elist = list + e * NT;

  // A-tile load assignment: one b128 per thread (64 rows x 4 chunks)
  int arow = tid >> 2;
  int achunk = tid & 3;
  int aslot = m_base + arow;
  int aslotc = (aslot < count) ? aslot : (count - 1);
  int atk = elist[aslotc];
  const unsigned short* Arow =
      Abuf + (size_t)((PHASE == 1) ? (atk >> 1) : atk) * KC;

  const unsigned short* Wte = Wt + (size_t)e * NC * KC;

  v8f acc[2][2];
  #pragma unroll
  for (int i = 0; i < 2; ++i)
    #pragma unroll
    for (int j = 0; j < 2; ++j)
      #pragma unroll
      for (int r = 0; r < 8; ++r) acc[i][j][r] = 0.0f;

  for (int k0 = 0; k0 < KC; k0 += 32) {
    // Stage A tile (gathered token rows)
    *(uint4*)(&sA[arow * 40 + achunk * 8]) =
        *(const uint4*)(Arow + k0 + achunk * 8);
    // Stage B tile (weights, [n][k] layout -> contiguous K)
    #pragma unroll
    for (int it = 0; it < 2; ++it) {
      int idx = tid + it * 256;
      int brow = idx >> 2;
      int bchunk = idx & 3;
      *(uint4*)(&sB[brow * 40 + bchunk * 8]) =
          *(const uint4*)(Wte + (size_t)(n_base + brow) * KC + k0 + bchunk * 8);
    }
    __syncthreads();

    #pragma unroll
    for (int mt = 0; mt < 2; ++mt) {
      // A fragment: 16-bit A 16x32 layout -> lane lm=M, chunks {8hi..}, {16+8hi..}
      int am = mw * 32 + mt * 16 + lm;
      union { uint4 q[2]; v16bf v; } ua;
      ua.q[0] = *(const uint4*)(&sA[am * 40 + hi * 8]);
      ua.q[1] = *(const uint4*)(&sA[am * 40 + 16 + hi * 8]);
      #pragma unroll
      for (int nt = 0; nt < 2; ++nt) {
        // B fragment: lane lm=N, chunks {8hi..}, {16+8hi..} (same K order as A)
        int bn = nw * 32 + nt * 16 + lm;
        union { uint4 q[2]; v16bf v; } ub;
        ub.q[0] = *(const uint4*)(&sB[bn * 40 + hi * 8]);
        ub.q[1] = *(const uint4*)(&sB[bn * 40 + 16 + hi * 8]);
        acc[mt][nt] = __builtin_amdgcn_wmma_f32_16x16x32_bf16(
            false, ua.v, false, ub.v, (short)0, acc[mt][nt], false, false);
      }
    }
    __syncthreads();
  }

  // Epilogue. C/D layout: element r -> m = r + 8*hi, n = lm.
  #pragma unroll
  for (int mt = 0; mt < 2; ++mt) {
    #pragma unroll
    for (int nt = 0; nt < 2; ++nt) {
      int n = n_base + nw * 32 + nt * 16 + lm;
      float bn = bias[e * NC + n];
      #pragma unroll
      for (int r = 0; r < 8; ++r) {
        int s = m_base + mw * 32 + mt * 16 + 8 * hi + r;
        if (s < count) {
          int tk = elist[s];
          float v = acc[mt][nt][r] + bn;
          if (PHASE == 1) {
            hout[(size_t)tk * HH + n] = f2bf(gelu_tanh(v));
          } else {
            int t = tk >> 1;
            atomicAdd(&out[(size_t)t * DM + n], gatew[tk] * v);
          }
        }
      }
    }
  }
}

// ---------------------------------------------------------------------------
// Workspace layout (bytes)
// ---------------------------------------------------------------------------
constexpr size_t OFF_CNT   = 0;                                   // 8 ints
constexpr size_t OFF_LIST  = 256;                                 // NE*NT ints
constexpr size_t OFF_GATEW = OFF_LIST + (size_t)NE * NT * 4;      // NT*2 f32
constexpr size_t OFF_X     = OFF_GATEW + (size_t)NT * TOPK * 4;   // NT*DM bf16
constexpr size_t OFF_W1T   = OFF_X + (size_t)NT * DM * 2;         // NE*HH*DM bf16
constexpr size_t OFF_W2T   = OFF_W1T + (size_t)NE * DM * HH * 2;  // NE*DM*HH bf16
constexpr size_t OFF_H     = OFF_W2T + (size_t)NE * HH * DM * 2;  // NT*2*HH bf16

extern "C" void kernel_launch(void* const* d_in, const int* in_sizes, int n_in,
                              void* d_out, int out_size, void* d_ws, size_t ws_size,
                              hipStream_t stream) {
  const float* inp = (const float*)d_in[0];
  const float* Wg  = (const float*)d_in[1];
  const float* bg  = (const float*)d_in[2];
  const float* W1  = (const float*)d_in[3];
  const float* b1  = (const float*)d_in[4];
  const float* W2  = (const float*)d_in[5];
  const float* b2  = (const float*)d_in[6];
  float* out = (float*)d_out;

  char* ws = (char*)d_ws;
  int*            cnt   = (int*)(ws + OFF_CNT);
  int*            list  = (int*)(ws + OFF_LIST);
  float*          gatew = (float*)(ws + OFF_GATEW);
  unsigned short* xb    = (unsigned short*)(ws + OFF_X);
  unsigned short* w1t   = (unsigned short*)(ws + OFF_W1T);
  unsigned short* w2t   = (unsigned short*)(ws + OFF_W2T);
  unsigned short* hb    = (unsigned short*)(ws + OFF_H);

  (void)in_sizes; (void)n_in; (void)out_size; (void)ws_size;

  int nxd = NT * DM;
  zero_kernel<<<(nxd + 255) / 256, 256, 0, stream>>>(out, nxd, cnt);
  conv_x_kernel<<<(nxd + 255) / 256, 256, 0, stream>>>(inp, xb, nxd);
  // W1 [NE][DM][HH] -> w1t [NE][HH][DM]
  transconv_kernel<<<dim3(HH / 32, DM / 32, NE), dim3(32, 8), 0, stream>>>(W1, w1t, DM, HH);
  // W2 [NE][HH][DM] -> w2t [NE][DM][HH]
  transconv_kernel<<<dim3(DM / 32, HH / 32, NE), dim3(32, 8), 0, stream>>>(W2, w2t, HH, DM);
  gate_kernel<<<NT / 8, 256, 0, stream>>>(inp, Wg, bg, cnt, list, gatew);
  expert_kernel<1><<<dim3(HH / 128, NT / 64, NE), 256, 0, stream>>>(
      xb, w1t, b1, cnt, list, gatew, hb, nullptr);
  expert_kernel<2><<<dim3(DM / 128, NT / 64, NE), 256, 0, stream>>>(
      hb, w2t, b2, cnt, list, gatew, nullptr, out);
}
